// ParallelLMHeadWithLoRA_53128745452281
// MI455X (gfx1250) — matrix operand order repaired and checked by reading, not compiled, not run on hardware
//
#include <hip/hip_runtime.h>
#include <hip/hip_bf16.h>

// ---------------------------------------------------------------------------
// Shapes (fixed by the reference)
// ---------------------------------------------------------------------------
#define N_TOK 2048
#define DIM   4096
#define VOC   32000
#define RANK  16

#define BM 128
#define BN 128
#define BK 32
#define LDB 40                    // padded LDS row stride in bf16 elems (80B)
#define KITERS (DIM / BK)         // 128

typedef __attribute__((ext_vector_type(16))) __bf16 v16bf;
typedef __attribute__((ext_vector_type(8)))  __bf16 v8bf;
typedef __attribute__((ext_vector_type(8)))  float  v8f;

// Pack two fp32 into (hi,hi) and (lo,lo) packed-bf16 dwords.
// hi = truncated top 16 bits (residual exact); lo = round-half-up bf16 of residual.
__device__ __forceinline__ void pack_pair(float a, float b,
                                          unsigned &hw, unsigned &lw) {
  unsigned ua = __float_as_uint(a), ub = __float_as_uint(b);
  hw = (ua >> 16) | (ub & 0xffff0000u);
  float ra = a - __uint_as_float(ua & 0xffff0000u);
  float rb = b - __uint_as_float(ub & 0xffff0000u);
  unsigned la = __float_as_uint(ra), lb = __float_as_uint(rb);
  lw = ((la + 0x8000u) >> 16) | ((lb + 0x8000u) & 0xffff0000u);
}

// ---------------------------------------------------------------------------
// Kernel 1: T[N_TOK, RANK] = H @ lora_A^T   (tiny: 2.7e8 FLOP)
// ---------------------------------------------------------------------------
__global__ __launch_bounds__(256)
void lora_a_gemv(const float* __restrict__ H, const float* __restrict__ A,
                 float* __restrict__ T) {
  const int gid = blockIdx.x * 256 + threadIdx.x;     // N_TOK*RANK threads
  const int n = gid >> 4;
  const int r = gid & 15;
  const float4* hp = (const float4*)(H + (size_t)n * DIM);
  const float4* ap = (const float4*)(A + (size_t)r * DIM);
  float s = 0.f;
  for (int i = 0; i < DIM / 4; ++i) {
    float4 hv = hp[i], av = ap[i];
    s = fmaf(hv.x, av.x, s); s = fmaf(hv.y, av.y, s);
    s = fmaf(hv.z, av.z, s); s = fmaf(hv.w, av.w, s);
  }
  T[gid] = s;
}

// ---------------------------------------------------------------------------
// Kernel 2: out = H @ W^T + T @ lora_B^T
//   128x128 block tile, 8 waves, each 64x32 (4x2 WMMA tiles).
//   LDS holds pre-split bf16 hi/lo planes; fragments are raw ds_load_b128
//   pairs. 3 bf16 WMMAs per K=32 chunk ~ fp32 accuracy at 3/8 the cost of the
//   native f32 WMMA path. Staging pipeline is one iteration deep:
//     iter kk: cvt_store(tile kk+1)  -> load_g(tile kk+2) -> compute(tile kk)
//   so global loads stay in flight across a full compute phase + barrier.
// ---------------------------------------------------------------------------
__global__ __launch_bounds__(256)
void lmhead_lora_gemm(const float* __restrict__ H, const float* __restrict__ W,
                      const float* __restrict__ T, const float* __restrict__ LB,
                      float* __restrict__ out) {
  __shared__ __bf16 sAh[2][BM * LDB];
  __shared__ __bf16 sAl[2][BM * LDB];
  __shared__ __bf16 sBh[2][BN * LDB];
  __shared__ __bf16 sBl[2][BN * LDB];

  const int tid  = threadIdx.x;
  const int wave = tid >> 5;
  const int lane = tid & 31;
  const int wm   = (wave & 1) * 64;   // wave row offset inside block tile
  const int wn   = (wave >> 1) * 32;  // wave col offset inside block tile
  const int l16  = lane & 15;
  const int lh   = lane >> 4;         // lane half (K-interleave select)

  const int m0 = blockIdx.x * BM;     // token-tile (fast varying for L2 W-reuse)
  const int n0 = blockIdx.y * BN;     // vocab-tile

  v8f acc[4][2];
#pragma unroll
  for (int mi = 0; mi < 4; ++mi)
#pragma unroll
    for (int ni = 0; ni < 2; ++ni) acc[mi][ni] = (v8f){};

  float4 rA[4], rB[4];                // register staging (tile kk+1 during iter kk)

  // 256 threads fetch a 128x32 fp32 tile as 1024 float4 (4 per thread).
  auto load_g = [&](float4* r, const float* __restrict__ src, int gr0, int k0) {
#pragma unroll
    for (int j = 0; j < 4; ++j) {
      const int idx = tid + 256 * j;
      const int row = idx >> 3;            // 8 float4 per row
      const int c4  = (idx & 7) << 2;
      r[j] = *(const float4*)(src + (size_t)(gr0 + row) * DIM + k0 + c4);
    }
  };

  // Split each staged float4 once into bf16 hi/lo planes (8B store each).
  auto cvt_store = [&](const float4* r, __bf16* dh, __bf16* dl) {
#pragma unroll
    for (int j = 0; j < 4; ++j) {
      const int idx = tid + 256 * j;
      const int row = idx >> 3;
      const int c4  = (idx & 7) << 2;
      uint2 hi, lo;
      pack_pair(r[j].x, r[j].y, hi.x, lo.x);
      pack_pair(r[j].z, r[j].w, hi.y, lo.y);
      *(uint2*)(dh + row * LDB + c4) = hi;
      *(uint2*)(dl + row * LDB + c4) = lo;
    }
  };

  // A fragment (16x32 bf16): lane m = l16; K(e) = (e>=8?16:0) + lh*8 + (e&7)
  auto frag_a = [&](const __bf16* plane, int mi) -> v16bf {
    const __bf16* p = plane + (wm + mi * 16 + l16) * LDB + lh * 8;
    v8bf a = *(const v8bf*)p;
    v8bf b = *(const v8bf*)(p + 16);
    return __builtin_shufflevector(a, b, 0, 1, 2, 3, 4, 5, 6, 7,
                                   8, 9, 10, 11, 12, 13, 14, 15);
  };

  // B fragment (32x16 bf16): lane n = l16; K(e) = (e>=8?16:0) + lh*8 + (e&7)
  auto frag_b = [&](const __bf16* plane, int ni) -> v16bf {
    const __bf16* p = plane + (wn + ni * 16 + l16) * LDB + lh * 8;
    v8bf a = *(const v8bf*)p;
    v8bf b = *(const v8bf*)(p + 16);
    return __builtin_shufflevector(a, b, 0, 1, 2, 3, 4, 5, 6, 7,
                                   8, 9, 10, 11, 12, 13, 14, 15);
  };

  // Prologue: tile 0 -> LDS buf0; tile 1 in flight in registers.
  load_g(rA, H, m0, 0);
  load_g(rB, W, n0, 0);
  cvt_store(rA, sAh[0], sAl[0]);
  cvt_store(rB, sBh[0], sBl[0]);
  load_g(rA, H, m0, BK);
  load_g(rB, W, n0, BK);
  __syncthreads();

  for (int kk = 0; kk < KITERS; ++kk) {
    const int cur = kk & 1;

    // Consume staged registers first (loads issued one iteration ago),
    // then put the next loads in flight for the entire compute phase.
    if (kk + 1 < KITERS) {
      cvt_store(rA, sAh[cur ^ 1], sAl[cur ^ 1]);
      cvt_store(rB, sBh[cur ^ 1], sBl[cur ^ 1]);
    }
    if (kk + 2 < KITERS) {
      load_g(rA, H, m0, (kk + 2) * BK);
      load_g(rB, W, n0, (kk + 2) * BK);
    }

    v16bf bh[2], bl[2];
#pragma unroll
    for (int ni = 0; ni < 2; ++ni) {
      bh[ni] = frag_b(sBh[cur], ni);
      bl[ni] = frag_b(sBl[cur], ni);
    }

    // Software-pipeline A fragments one mi ahead of the WMMA group.
    v16bf ah = frag_a(sAh[cur], 0);
    v16bf al = frag_a(sAl[cur], 0);
#pragma unroll
    for (int mi = 0; mi < 4; ++mi) {
      v16bf ahn, aln;
      if (mi < 3) {
        ahn = frag_a(sAh[cur], mi + 1);
        aln = frag_a(sAl[cur], mi + 1);
      }
#pragma unroll
      for (int ni = 0; ni < 2; ++ni) {
        // fp32 ~ (Ah+Al)(Bh+Bl): keep the three dominant bf16 products
        acc[mi][ni] = __builtin_amdgcn_wmma_f32_16x16x32_bf16(
            false, ah, false, bh[ni], (short)0, acc[mi][ni], false, false);
        acc[mi][ni] = __builtin_amdgcn_wmma_f32_16x16x32_bf16(
            false, ah, false, bl[ni], (short)0, acc[mi][ni], false, false);
        acc[mi][ni] = __builtin_amdgcn_wmma_f32_16x16x32_bf16(
            false, al, false, bh[ni], (short)0, acc[mi][ni], false, false);
      }
      ah = ahn;
      al = aln;
    }
    __syncthreads();                  // readers done + next tiles visible
  }

  // Epilogue: fold in LoRA correction acc += T[m,:16] . lora_B[n,:16], store NT.
  float brow[2][16];
#pragma unroll
  for (int ni = 0; ni < 2; ++ni) {
    const float* bp = LB + (size_t)(n0 + wn + ni * 16 + l16) * RANK;
#pragma unroll
    for (int e = 0; e < 16; ++e) brow[ni][e] = bp[e];
  }

#pragma unroll
  for (int mi = 0; mi < 4; ++mi) {
#pragma unroll
    for (int r = 0; r < 8; ++r) {
      const int m = m0 + wm + mi * 16 + lh * 8 + r;   // C layout: lanes 16-31 -> M+8
      const float* tp = T + (size_t)m * RANK;
      float t[16];
#pragma unroll
      for (int e = 0; e < 16; ++e) t[e] = tp[e];
#pragma unroll
      for (int ni = 0; ni < 2; ++ni) {
        float corr = 0.f;
#pragma unroll
        for (int e = 0; e < 16; ++e) corr = fmaf(t[e], brow[ni][e], corr);
        const int n = n0 + wn + ni * 16 + l16;
        __builtin_nontemporal_store(acc[mi][ni][r] + corr,
                                    out + (size_t)m * VOC + n);
      }
    }
  }
}

// ---------------------------------------------------------------------------
extern "C" void kernel_launch(void* const* d_in, const int* in_sizes, int n_in,
                              void* d_out, int out_size, void* d_ws, size_t ws_size,
                              hipStream_t stream) {
  (void)in_sizes; (void)n_in; (void)out_size; (void)ws_size;
  const float* H  = (const float*)d_in[0];   // [2048, 4096]
  const float* W  = (const float*)d_in[1];   // [32000, 4096]
  const float* A  = (const float*)d_in[2];   // [16, 4096]
  const float* LB = (const float*)d_in[3];   // [32000, 16]
  float* out = (float*)d_out;                // [2048, 32000]
  float* T   = (float*)d_ws;                 // [2048, 16] scratch

  lora_a_gemv<<<(N_TOK * RANK) / 256, 256, 0, stream>>>(H, A, T);

  dim3 grid(N_TOK / BM, VOC / BN);           // x = M tile (fast) for L2 W-reuse
  lmhead_lora_gemm<<<grid, 256, 0, stream>>>(H, W, T, LB, out);
}
